// TrajectoryImitationRNNModel_44916767981651
// MI455X (gfx1250) — compile-verified, matrix operand order checked by reading, not run on hardware
//
#include <hip/hip_runtime.h>
#include <hip/hip_bf16.h>

// ---------------------------------------------------------------------------
// Types for CDNA5 WMMA (wave32): v_wmma_f32_16x16x32_bf16
// ---------------------------------------------------------------------------
typedef __attribute__((ext_vector_type(16))) __bf16 v16bf;
typedef __attribute__((ext_vector_type(8)))  float  v8f;

__device__ __forceinline__ unsigned short f2bf(float f) {
  unsigned int u = __float_as_uint(f);
  u += 0x7FFFu + ((u >> 16) & 1u);          // round-to-nearest-even
  return (unsigned short)(u >> 16);
}

// ---------------------------------------------------------------------------
// f32 -> bf16 conversion with zero padding: src [rowsSrc, colsSrc] ->
// dst [rowsDst, colsDst] (row-major). Used for both weights and activations.
// ---------------------------------------------------------------------------
__global__ __launch_bounds__(256)
void cvt_pad_bf16(const float* __restrict__ src, unsigned short* __restrict__ dst,
                  int rowsSrc, int colsSrc, int rowsDst, int colsDst) {
  size_t i = (size_t)blockIdx.x * blockDim.x + threadIdx.x;
  size_t total = (size_t)rowsDst * colsDst;
  if (i >= total) return;
  int r = (int)(i / colsDst), c = (int)(i % colsDst);
  float v = (r < rowsSrc && c < colsSrc) ? src[(size_t)r * colsSrc + c] : 0.f;
  dst[i] = f2bf(v);
}

// Build bf16 Fcat = [img_enc (1000) | P3 (256) | zeros (24)]  -> [32,1280]
__global__ __launch_bounds__(256)
void cvt_concat_bf16(const float* __restrict__ imgenc, const float* __restrict__ P3,
                     unsigned short* __restrict__ dst) {
  int i = blockIdx.x * blockDim.x + threadIdx.x;
  if (i >= 32 * 1280) return;
  int b = i / 1280, c = i % 1280;
  float v = 0.f;
  if (c < 1000)      v = imgenc[b * 1000 + c];
  else if (c < 1256) v = P3[b * 256 + (c - 1000)];
  dst[i] = f2bf(v);
}

// ---------------------------------------------------------------------------
// Skinny GEMM, M = 32 fixed.  out[m][n] = act(sum_k A[m][k] * W[n][k] + bias[n])
// A: [32, Kp] bf16 (K zero-padded to mult of 32)
// B: [Np, Kp] bf16 (weights row-major [n][k], N zero-padded to mult of 16)
// One wave per 16x16 output tile; v_wmma_f32_16x16x32_bf16 over the K loop.
// ISA 16-bit A/B fragment layout: lane&15 = row, lane>=16 selects K+8 half,
// dword v(0..3) holds K pairs {2v,2v+1}, v(4..7) the +16 half.
// ---------------------------------------------------------------------------
__global__ __launch_bounds__(128)
void gemm_bf16_wmma(const unsigned short* __restrict__ A,
                    const unsigned short* __restrict__ B,
                    const float* __restrict__ bias,
                    float* __restrict__ C,
                    int Kp, int Nact, int ntilesN, int ldC, int relu) {
  int wave = threadIdx.x >> 5;
  int lane = threadIdx.x & 31;
  int tile = blockIdx.x * 4 + wave;
  if (tile >= 2 * ntilesN) return;
  int mt = tile & 1;
  int nt = tile >> 1;

  int rsel  = lane & 15;                 // row-in-tile for A (m) and B (n)
  int kdofs = (lane >> 4) << 2;          // dword offset for K half: 0 or 4 (=K+8)

  const unsigned int* Arow = (const unsigned int*)(A + (size_t)(mt * 16 + rsel) * Kp);
  const unsigned int* Brow = (const unsigned int*)(B + (size_t)(nt * 16 + rsel) * Kp);

  v8f acc = {};
  union FragA { v16bf v; unsigned int u[8]; } af;
  union FragB { v16bf v; unsigned int u[8]; } bf;

  for (int k0 = 0; k0 < Kp; k0 += 32) {
    int dbase = (k0 >> 1) + kdofs;
#pragma unroll
    for (int v = 0; v < 8; ++v) {
      int d = dbase + (v & 3) + ((v >> 2) << 3);
      af.u[v] = Arow[d];
      bf.u[v] = Brow[d];
    }
    if (k0 + 32 < Kp) {                   // prefetch next K-chunk (global_prefetch_b8)
      __builtin_prefetch(Brow + dbase + 16, 0, 1);
      __builtin_prefetch(Arow + dbase + 16, 0, 1);
    }
    acc = __builtin_amdgcn_wmma_f32_16x16x32_bf16(
        false, af.v, false, bf.v, (short)0, acc, false, false);
  }

  int n = nt * 16 + rsel;
  if (n < Nact) {
    float bv = bias ? bias[n] : 0.f;
    int mbase = mt * 16 + ((lane >> 4) << 3);   // C layout: VGPR j -> M = j (+8 for hi lanes)
#pragma unroll
    for (int j = 0; j < 8; ++j) {
      float r = acc[j] + bv;
      if (relu) r = fmaxf(r, 0.f);
      C[(size_t)(mbase + j) * ldC + n] = r;
    }
  }
}

// ---------------------------------------------------------------------------
// conv(X[32,12,200,200], comp_W[3,12,3,3], pad=1) -> x3 [32,3,200,200]  (no relu)
// ---------------------------------------------------------------------------
__global__ __launch_bounds__(256)
void conv_comp_kernel(const float* __restrict__ X, const float* __restrict__ W,
                      const float* __restrict__ bias, float* __restrict__ out) {
  int i = blockIdx.x * blockDim.x + threadIdx.x;
  if (i >= 32 * 3 * 40000) return;
  int b = i / 120000; int r = i % 120000;
  int oc = r / 40000;  int p = r % 40000;
  int oy = p / 200, ox = p % 200;
  const float* xb = X + (size_t)b * 12 * 40000;
  float acc = bias[oc];
  for (int ic = 0; ic < 12; ++ic) {
    const float* xc = xb + ic * 40000;
    const float* wc = W + (oc * 12 + ic) * 9;
#pragma unroll
    for (int ky = 0; ky < 3; ++ky) {
      int iy = oy + ky - 1; if ((unsigned)iy >= 200u) continue;
#pragma unroll
      for (int kx = 0; kx < 3; ++kx) {
        int ix = ox + kx - 1; if ((unsigned)ix >= 200u) continue;
        acc += xc[iy * 200 + ix] * wc[ky * 3 + kx];
      }
    }
  }
  out[i] = acc;
}

// ---------------------------------------------------------------------------
// relu(conv(x3, feat_W[32,3,3,3], stride=2, pad=1)) fused with global-avg-pool.
// One block per (b, oc); LDS tree reduction over the 100x100 outputs.
// ---------------------------------------------------------------------------
__global__ __launch_bounds__(256)
void conv_feat_gap_kernel(const float* __restrict__ x3, const float* __restrict__ W,
                          const float* __restrict__ bias, float* __restrict__ gap) {
  int b  = blockIdx.x >> 5;
  int oc = blockIdx.x & 31;
  const float* xin = x3 + (size_t)b * 3 * 40000;
  float partial = 0.f;
  for (int p = threadIdx.x; p < 10000; p += 256) {
    int oy = p / 100, ox = p % 100;
    float acc = bias[oc];
    for (int ic = 0; ic < 3; ++ic) {
      const float* xc = xin + ic * 40000;
      const float* wc = W + (oc * 3 + ic) * 9;
#pragma unroll
      for (int ky = 0; ky < 3; ++ky) {
        int iy = oy * 2 + ky - 1; if ((unsigned)iy >= 200u) continue;
#pragma unroll
        for (int kx = 0; kx < 3; ++kx) {
          int ix = ox * 2 + kx - 1; if ((unsigned)ix >= 200u) continue;
          acc += xc[iy * 200 + ix] * wc[ky * 3 + kx];
        }
      }
    }
    partial += fmaxf(acc, 0.f);
  }
  __shared__ float s[256];
  s[threadIdx.x] = partial;
  __syncthreads();
  for (int st = 128; st > 0; st >>= 1) {
    if (threadIdx.x < st) s[threadIdx.x] += s[threadIdx.x + st];
    __syncthreads();
  }
  if (threadIdx.x == 0) gap[b * 32 + oc] = s[0] * (1.f / 10000.f);
}

// ---------------------------------------------------------------------------
// relu(conv(M[32,2,200,200], mem_W[1,2,121,121], pad=10)) -> P1 [32,10000]
// ---------------------------------------------------------------------------
__global__ __launch_bounds__(256)
void mem_conv_kernel(const float* __restrict__ M, const float* __restrict__ W,
                     const float* __restrict__ bias, float* __restrict__ out) {
  int i = blockIdx.x * blockDim.x + threadIdx.x;
  if (i >= 32 * 10000) return;
  int b = i / 10000; int p = i % 10000;
  int oy = p / 100, ox = p % 100;
  const float* M0 = M + (size_t)b * 80000;
  const float* M1 = M0 + 40000;
  float acc = bias[0];
  for (int ky = 0; ky < 121; ++ky) {
    int iy = oy + ky - 10; if ((unsigned)iy >= 200u) continue;
    const float* r0 = M0 + iy * 200;
    const float* r1 = M1 + iy * 200;
    const float* w0 = W + ky * 121;
    const float* w1 = w0 + 14641;
    for (int kx = 0; kx < 121; ++kx) {
      int ix = ox + kx - 10; if ((unsigned)ix >= 200u) continue;
      acc += w0[kx] * r0[ix] + w1[kx] * r1[ix];
    }
  }
  out[i] = fmaxf(acc, 0.f);
}

// ---------------------------------------------------------------------------
// relu(ConvTranspose2d(F5[32,1,16,16], ct_W[1,2,70,70], stride=10, pad=10))
//   -> F [32,2,200,200].  ky = oy+10-10*iy  (<=7x7 taps per output pixel).
// ---------------------------------------------------------------------------
__global__ __launch_bounds__(256)
void convt_kernel(const float* __restrict__ Fin, const float* __restrict__ W,
                  const float* __restrict__ bias, float* __restrict__ Fout) {
  int i = blockIdx.x * blockDim.x + threadIdx.x;
  if (i >= 32 * 2 * 40000) return;
  int b = i / 80000; int r = i % 80000;
  int oc = r / 40000; int p = r % 40000;
  int oy = p / 200, ox = p % 200;
  int sy = oy + 10, sx = ox + 10;
  int iy0 = (sy >= 69) ? (sy - 69 + 9) / 10 : 0;
  int iy1 = sy / 10; if (iy1 > 15) iy1 = 15;
  int ix0 = (sx >= 69) ? (sx - 69 + 9) / 10 : 0;
  int ix1 = sx / 10; if (ix1 > 15) ix1 = 15;
  const float* fin = Fin + b * 256;
  const float* w   = W + oc * 4900;
  float acc = bias[oc];
  for (int iy = iy0; iy <= iy1; ++iy) {
    int ky = sy - 10 * iy;
    for (int ix = ix0; ix <= ix1; ++ix) {
      int kx = sx - 10 * ix;
      acc += fin[iy * 16 + ix] * w[ky * 70 + kx];
    }
  }
  Fout[i] = fmaxf(acc, 0.f);
}

// ---------------------------------------------------------------------------
// Per batch row: stable softmax + first-index argmax over F[:,0] (40000),
// sigmoid over F[:,1]. One 1024-thread block (32 waves) per b, wave32 shuffles.
// ---------------------------------------------------------------------------
__global__ __launch_bounds__(1024)
void softmax_argmax_sigmoid_kernel(const float* __restrict__ F,
                                   float* __restrict__ outP, float* __restrict__ outB,
                                   int* __restrict__ idxOut, int t) {
  int b = blockIdx.x;
  const float* f0 = F + (size_t)b * 80000;
  const float* f1 = f0 + 40000;
  float* oP = outP + ((size_t)b * 10 + t) * 40000;
  float* oB = outB + ((size_t)b * 10 + t) * 40000;
  __shared__ float sm[32];
  __shared__ int   si[32];
  __shared__ float ss[32];
  int wid = threadIdx.x >> 5, lane = threadIdx.x & 31;

  float m = -3.4e38f; int mi = 0x7fffffff;
  for (int j = threadIdx.x; j < 40000; j += 1024) {
    float v = f0[j];
    if (v > m || (v == m && j < mi)) { m = v; mi = j; }
  }
  for (int o = 16; o > 0; o >>= 1) {
    float om = __shfl_down(m, o, 32);
    int   oi = __shfl_down(mi, o, 32);
    if (om > m || (om == m && oi < mi)) { m = om; mi = oi; }
  }
  if (lane == 0) { sm[wid] = m; si[wid] = mi; }
  __syncthreads();
  if (wid == 0) {
    m = sm[lane]; mi = si[lane];
    for (int o = 16; o > 0; o >>= 1) {
      float om = __shfl_down(m, o, 32);
      int   oi = __shfl_down(mi, o, 32);
      if (om > m || (om == m && oi < mi)) { m = om; mi = oi; }
    }
    if (lane == 0) { sm[0] = m; si[0] = mi; idxOut[b] = mi; }
  }
  __syncthreads();
  m = sm[0];

  float sum = 0.f;
  for (int j = threadIdx.x; j < 40000; j += 1024) sum += __expf(f0[j] - m);
  for (int o = 16; o > 0; o >>= 1) sum += __shfl_down(sum, o, 32);
  if (lane == 0) ss[wid] = sum;
  __syncthreads();
  if (wid == 0) {
    sum = ss[lane];
    for (int o = 16; o > 0; o >>= 1) sum += __shfl_down(sum, o, 32);
    if (lane == 0) ss[0] = sum;
  }
  __syncthreads();
  float inv = 1.f / ss[0];
  for (int j = threadIdx.x; j < 40000; j += 1024) {
    oP[j] = __expf(f0[j] - m) * inv;
    float v = f1[j];
    oB[j] = 1.f / (1.f + __expf(-v));
  }
}

// ---------------------------------------------------------------------------
// relu(conv(F[32,2,200,200], oc_W[1,2,3,3], pad=1)) -> pp [32,40000]
// ---------------------------------------------------------------------------
__global__ __launch_bounds__(256)
void conv_oc_kernel(const float* __restrict__ F, const float* __restrict__ W,
                    const float* __restrict__ bias, float* __restrict__ pp) {
  int i = blockIdx.x * blockDim.x + threadIdx.x;
  if (i >= 32 * 40000) return;
  int b = i / 40000; int p = i % 40000;
  int oy = p / 200, ox = p % 200;
  const float* fb = F + (size_t)b * 80000;
  float acc = bias[0];
  for (int c = 0; c < 2; ++c) {
    const float* fc = fb + c * 40000;
    const float* wc = W + c * 9;
#pragma unroll
    for (int ky = 0; ky < 3; ++ky) {
      int iy = oy + ky - 1; if ((unsigned)iy >= 200u) continue;
#pragma unroll
      for (int kx = 0; kx < 3; ++kx) {
        int ix = ox + kx - 1; if ((unsigned)ix >= 200u) continue;
        acc += fc[iy * 200 + ix] * wc[ky * 3 + kx];
      }
    }
  }
  pp[i] = fmaxf(acc, 0.f);
}

// ---------------------------------------------------------------------------
// Recurrent state helpers
// ---------------------------------------------------------------------------
__global__ __launch_bounds__(256)
void m_init_kernel(const float* __restrict__ M_B_0, float* __restrict__ M) {
  int i = blockIdx.x * blockDim.x + threadIdx.x;
  if (i >= 32 * 80000) return;
  M[i] = M_B_0[i % 80000];
}

__global__ __launch_bounds__(256)
void m_copy_ch1_kernel(const float* __restrict__ F, float* __restrict__ M) {
  int i = blockIdx.x * blockDim.x + threadIdx.x;
  if (i >= 32 * 40000) return;
  int b = i / 40000, p = i % 40000;
  M[(size_t)b * 80000 + 40000 + p] = F[(size_t)b * 80000 + 40000 + p];
}

// Faithful to the source bug: every batch item gets ALL batch argmax cells = 1.
__global__ __launch_bounds__(1024)
void m_set_points_kernel(float* __restrict__ M, const int* __restrict__ idx) {
  int i = threadIdx.x;
  if (i >= 1024) return;
  int b = i >> 5, j = i & 31;
  M[(size_t)b * 80000 + idx[j]] = 1.f;
}

// ---------------------------------------------------------------------------
// Host driver
// ---------------------------------------------------------------------------
static inline unsigned cdiv256(size_t n) { return (unsigned)((n + 255) / 256); }

extern "C" void kernel_launch(void* const* d_in, const int* in_sizes, int n_in,
                              void* d_out, int out_size, void* d_ws, size_t ws_size,
                              hipStream_t stream) {
  const float* X        = (const float*)d_in[0];
  const float* comp_W   = (const float*)d_in[1];
  const float* comp_b   = (const float*)d_in[2];
  const float* feat_W   = (const float*)d_in[3];
  const float* feat_b   = (const float*)d_in[4];
  const float* featfc_W = (const float*)d_in[5];
  const float* featfc_b = (const float*)d_in[6];
  const float* M_B_0    = (const float*)d_in[7];
  const float* mem_W    = (const float*)d_in[8];
  const float* mem_b    = (const float*)d_in[9];
  const float* mf_W1    = (const float*)d_in[10];
  const float* mf_b1    = (const float*)d_in[11];
  const float* mf_W2    = (const float*)d_in[12];
  const float* mf_b2    = (const float*)d_in[13];
  const float* in_W1    = (const float*)d_in[14];
  const float* in_b1    = (const float*)d_in[15];
  const float* in_W2    = (const float*)d_in[16];
  const float* in_b2    = (const float*)d_in[17];
  const float* ct_W     = (const float*)d_in[18];
  const float* ct_b     = (const float*)d_in[19];
  const float* oc_W     = (const float*)d_in[20];
  const float* oc_b     = (const float*)d_in[21];
  const float* of_W1    = (const float*)d_in[22];
  const float* of_b1    = (const float*)d_in[23];
  const float* of_W2    = (const float*)d_in[24];
  const float* of_b2    = (const float*)d_in[25];
  const float* of_W3    = (const float*)d_in[26];
  const float* of_b3    = (const float*)d_in[27];

  char* ws = (char*)d_ws;
  size_t off = 0;
  auto bump = [&](size_t bytes) -> void* {
    void* p = (void*)(ws + off);
    off = (off + bytes + 255) & ~(size_t)255;
    return p;
  };

  // f32 scratch
  float* Mbuf   = (float*)bump((size_t)32 * 80000 * 4);
  float* Fbuf   = (float*)bump((size_t)32 * 80000 * 4);
  float* x3     = (float*)bump((size_t)32 * 3 * 40000 * 4);
  float* gapf   = (float*)bump((size_t)32 * 32 * 4);
  float* imgenc = (float*)bump((size_t)32 * 1000 * 4);
  float* P1f    = (float*)bump((size_t)32 * 10000 * 4);
  float* P2f    = (float*)bump((size_t)32 * 500 * 4);
  float* P3f    = (float*)bump((size_t)32 * 256 * 4);
  float* F4f    = (float*)bump((size_t)32 * 500 * 4);
  float* F5f    = (float*)bump((size_t)32 * 256 * 4);
  float* ppf    = (float*)bump((size_t)32 * 40000 * 4);
  float* q1f    = (float*)bump((size_t)32 * 1000 * 4);
  float* q2f    = (float*)bump((size_t)32 * 120 * 4);
  int*   idxb   = (int*)  bump((size_t)32 * 4);

  // bf16 scratch (ushort storage)
  unsigned short* gapb  = (unsigned short*)bump((size_t)32 * 32 * 2);
  unsigned short* Wfcb  = (unsigned short*)bump((size_t)1008 * 32 * 2);
  unsigned short* P1b   = (unsigned short*)bump((size_t)32 * 10016 * 2);
  unsigned short* Wmf1b = (unsigned short*)bump((size_t)512 * 10016 * 2);
  unsigned short* P2b   = (unsigned short*)bump((size_t)32 * 512 * 2);
  unsigned short* Wmf2b = (unsigned short*)bump((size_t)256 * 512 * 2);
  unsigned short* Fcatb = (unsigned short*)bump((size_t)32 * 1280 * 2);
  unsigned short* Win1b = (unsigned short*)bump((size_t)512 * 1280 * 2);
  unsigned short* F4b   = (unsigned short*)bump((size_t)32 * 512 * 2);
  unsigned short* Win2b = (unsigned short*)bump((size_t)256 * 512 * 2);
  unsigned short* ppb   = (unsigned short*)bump((size_t)32 * 40000 * 2);
  unsigned short* Wof1b = (unsigned short*)bump((size_t)1008 * 40000 * 2);
  unsigned short* q1b   = (unsigned short*)bump((size_t)32 * 1024 * 2);
  unsigned short* Wof2b = (unsigned short*)bump((size_t)128 * 1024 * 2);
  unsigned short* q2b   = (unsigned short*)bump((size_t)32 * 128 * 2);
  unsigned short* Wof3b = (unsigned short*)bump((size_t)16 * 128 * 2);

  float* out_pos = (float*)d_out;                    // [32,10,1,200,200]
  float* out_box = out_pos + (size_t)32 * 10 * 40000;
  float* out_pts = out_box + (size_t)32 * 10 * 40000;  // [32,10,4]

  // ---- one-time: weight conversion to bf16 (padded) ----
  cvt_pad_bf16<<<cdiv256((size_t)1008 * 32), 256, 0, stream>>>(featfc_W, Wfcb, 1000, 32, 1008, 32);
  cvt_pad_bf16<<<cdiv256((size_t)512 * 10016), 256, 0, stream>>>(mf_W1, Wmf1b, 500, 10000, 512, 10016);
  cvt_pad_bf16<<<cdiv256((size_t)256 * 512), 256, 0, stream>>>(mf_W2, Wmf2b, 256, 500, 256, 512);
  cvt_pad_bf16<<<cdiv256((size_t)512 * 1280), 256, 0, stream>>>(in_W1, Win1b, 500, 1256, 512, 1280);
  cvt_pad_bf16<<<cdiv256((size_t)256 * 512), 256, 0, stream>>>(in_W2, Win2b, 256, 500, 256, 512);
  cvt_pad_bf16<<<cdiv256((size_t)1008 * 40000), 256, 0, stream>>>(of_W1, Wof1b, 1000, 40000, 1008, 40000);
  cvt_pad_bf16<<<cdiv256((size_t)128 * 1024), 256, 0, stream>>>(of_W2, Wof2b, 120, 1000, 128, 1024);
  cvt_pad_bf16<<<cdiv256((size_t)16 * 128), 256, 0, stream>>>(of_W3, Wof3b, 4, 120, 16, 128);

  // ---- one-time: image encoder ----
  m_init_kernel<<<cdiv256((size_t)32 * 80000), 256, 0, stream>>>(M_B_0, Mbuf);
  conv_comp_kernel<<<cdiv256((size_t)32 * 3 * 40000), 256, 0, stream>>>(X, comp_W, comp_b, x3);
  conv_feat_gap_kernel<<<1024, 256, 0, stream>>>(x3, feat_W, feat_b, gapf);
  cvt_pad_bf16<<<cdiv256((size_t)32 * 32), 256, 0, stream>>>(gapf, gapb, 32, 32, 32, 32);
  gemm_bf16_wmma<<<(2 * 63 + 3) / 4, 128, 0, stream>>>(gapb, Wfcb, featfc_b, imgenc,
                                                       32, 1000, 63, 1000, 0);

  // ---- recurrent steps ----
  for (int t = 0; t < 10; ++t) {
    mem_conv_kernel<<<cdiv256((size_t)32 * 10000), 256, 0, stream>>>(Mbuf, mem_W, mem_b, P1f);
    cvt_pad_bf16<<<cdiv256((size_t)32 * 10016), 256, 0, stream>>>(P1f, P1b, 32, 10000, 32, 10016);
    gemm_bf16_wmma<<<(2 * 32 + 3) / 4, 128, 0, stream>>>(P1b, Wmf1b, mf_b1, P2f,
                                                         10016, 500, 32, 500, 1);
    cvt_pad_bf16<<<cdiv256((size_t)32 * 512), 256, 0, stream>>>(P2f, P2b, 32, 500, 32, 512);
    gemm_bf16_wmma<<<(2 * 16 + 3) / 4, 128, 0, stream>>>(P2b, Wmf2b, mf_b2, P3f,
                                                         512, 256, 16, 256, 1);
    cvt_concat_bf16<<<cdiv256((size_t)32 * 1280), 256, 0, stream>>>(imgenc, P3f, Fcatb);
    gemm_bf16_wmma<<<(2 * 32 + 3) / 4, 128, 0, stream>>>(Fcatb, Win1b, in_b1, F4f,
                                                         1280, 500, 32, 500, 1);
    cvt_pad_bf16<<<cdiv256((size_t)32 * 512), 256, 0, stream>>>(F4f, F4b, 32, 500, 32, 512);
    gemm_bf16_wmma<<<(2 * 16 + 3) / 4, 128, 0, stream>>>(F4b, Win2b, in_b2, F5f,
                                                         512, 256, 16, 256, 1);
    convt_kernel<<<cdiv256((size_t)32 * 2 * 40000), 256, 0, stream>>>(F5f, ct_W, ct_b, Fbuf);
    softmax_argmax_sigmoid_kernel<<<32, 1024, 0, stream>>>(Fbuf, out_pos, out_box, idxb, t);
    conv_oc_kernel<<<cdiv256((size_t)32 * 40000), 256, 0, stream>>>(Fbuf, oc_W, oc_b, ppf);
    cvt_pad_bf16<<<cdiv256((size_t)32 * 40000), 256, 0, stream>>>(ppf, ppb, 32, 40000, 32, 40000);
    gemm_bf16_wmma<<<(2 * 63 + 3) / 4, 128, 0, stream>>>(ppb, Wof1b, of_b1, q1f,
                                                         40000, 1000, 63, 1000, 1);
    cvt_pad_bf16<<<cdiv256((size_t)32 * 1024), 256, 0, stream>>>(q1f, q1b, 32, 1000, 32, 1024);
    gemm_bf16_wmma<<<(2 * 8 + 3) / 4, 128, 0, stream>>>(q1b, Wof2b, of_b2, q2f,
                                                        1024, 120, 8, 120, 1);
    cvt_pad_bf16<<<cdiv256((size_t)32 * 128), 256, 0, stream>>>(q2f, q2b, 32, 120, 32, 128);
    // Final 120->4: write straight into pred_points[b, t, :] (row stride 40)
    gemm_bf16_wmma<<<1, 128, 0, stream>>>(q2b, Wof3b, of_b3, out_pts + (size_t)t * 4,
                                          128, 4, 1, 40, 1);
    // State update: ch1 <- relu convT ch1 (pre-sigmoid); ch0 gets all argmax cells = 1
    m_copy_ch1_kernel<<<cdiv256((size_t)32 * 40000), 256, 0, stream>>>(Fbuf, Mbuf);
    m_set_points_kernel<<<1, 1024, 0, stream>>>(Mbuf, idxb);
  }
}
